// SwinTransformerBlock_27307402068325
// MI455X (gfx1250) — matrix operand order repaired and checked by reading, not compiled, not run on hardware
//
#include <hip/hip_runtime.h>
#include <hip/hip_bf16.h>
#include <math.h>

typedef __bf16 bf16;
typedef __attribute__((ext_vector_type(16))) __bf16 v16bf;
typedef __attribute__((ext_vector_type(8)))  __bf16 v8bf;
typedef __attribute__((ext_vector_type(8)))  float   v8f;

// gcc-style int4 vector to match the async-LDS builtin's parameter type
typedef int v4i __attribute__((vector_size(16)));
typedef __attribute__((address_space(1))) v4i as1_v4i;
typedef __attribute__((address_space(3))) v4i as3_v4i;

#if defined(__has_builtin)
#if __has_builtin(__builtin_amdgcn_global_load_async_to_lds_b128) && \
    __has_builtin(__builtin_amdgcn_s_wait_asynccnt)
#define HAVE_ASYNC_LDS 1
#endif
#endif
#ifndef HAVE_ASYNC_LDS
#define HAVE_ASYNC_LDS 0
#endif

#define C_      384
#define HEADS_  12
#define HD_     32
#define NTOK    49          // tokens per window (7x7)
#define NROWS   100352      // 32 * 3136 total tokens
#define MLPH    1536
#define QKV_N   1152

// ---------------------------------------------------------------------------
// window-row -> image-row mapping (includes the cyclic shift by 3).
// Used both for gather (LN1 input) and scatter (proj output); it is a bijection.
__device__ __forceinline__ int win_to_img_row(int r) {
    int w    = r / NTOK;
    int tok  = r - w * NTOK;
    int b    = w >> 6;          // 64 windows per image
    int widx = w & 63;
    int wi   = widx >> 3, wj = widx & 7;
    int ti   = tok / 7,   tj = tok - ti * 7;
    int hs = wi * 7 + ti;
    int ws = wj * 7 + tj;
    int h = hs + 3; if (h >= 56) h -= 56;
    int ww = ws + 3; if (ww >= 56) ww -= 56;
    return b * 3136 + h * 56 + ww;
}

// 16-bit WMMA A-fragment, contiguous k order:
// lane holds K = {c0..c0+15} with c0 = k0 + half*16.
__device__ __forceinline__ v16bf load_a_frag(const bf16* rowp, int c0) {
    v8bf lo = *(const v8bf*)(rowp + c0);
    v8bf hi = *(const v8bf*)(rowp + c0 + 8);
    v16bf a;
#pragma unroll
    for (int i = 0; i < 8; ++i) { a[i] = lo[i]; a[8 + i] = hi[i]; }
    return a;
}

// ---------------------------------------------------------------------------
__global__ void pack_bf16(const float* __restrict__ s, bf16* __restrict__ d, int n) {
    int i = blockIdx.x * 256 + threadIdx.x;
    if (i < n) d[i] = (bf16)s[i];
}

// ---------------------------------------------------------------------------
// Precompute combined (relative-position bias + shift mask) tables.
// T[h][cls][64][64], cls = 2*(wi==7) + (wj==7). Padded entries get -1e30 so the
// attention epilogue needs no clamping / region math at all.
__global__ __launch_bounds__(256)
void build_attn_tables(const float* __restrict__ rb, float* __restrict__ T) {
    int h   = blockIdx.x >> 2;
    int cls = blockIdx.x & 3;
    int wi7 = cls >> 1, wj7 = cls & 1;
    for (int i = threadIdx.x; i < 64 * 64; i += 256) {
        int m = i >> 6, n = i & 63;
        float v = -1e30f;
        if (m < NTOK && n < NTOK) {
            int mi = m / 7, mj = m - mi * 7;
            int ni = n / 7, nj = n - ni * 7;
            float bias = rb[h * 169 + (mi - ni + 6) * 13 + (mj - nj + 6)];
            int cm = 3 * (wi7 ? (mi < 4 ? 1 : 2) : 0) + (wj7 ? (mj < 4 ? 1 : 2) : 0);
            int cn = 3 * (wi7 ? (ni < 4 ? 1 : 2) : 0) + (wj7 ? (nj < 4 ? 1 : 2) : 0);
            v = bias + ((cm != cn) ? -100.0f : 0.0f);
        }
        T[(size_t)blockIdx.x * 4096 + i] = v;
    }
}

// ---------------------------------------------------------------------------
// LayerNorm; WINMAP=true gathers rows through shift+window-partition map.
template<bool WINMAP>
__global__ __launch_bounds__(256)
void ln_kernel(const float* __restrict__ x, const float* __restrict__ w,
               const float* __restrict__ b, bf16* __restrict__ out) {
    int wave = threadIdx.x >> 5, lane = threadIdx.x & 31;
    int r = blockIdx.x * 8 + wave;
    int src_row = WINMAP ? win_to_img_row(r) : r;
    const float* src = x + (size_t)src_row * C_;
    float vals[12], s = 0.f, s2 = 0.f;
#pragma unroll
    for (int i = 0; i < 12; ++i) {
        float v = src[lane + i * 32];
        vals[i] = v; s += v; s2 += v * v;
    }
#pragma unroll
    for (int m = 16; m >= 1; m >>= 1) {
        s  += __shfl_xor(s,  m, 32);
        s2 += __shfl_xor(s2, m, 32);
    }
    float mu  = s * (1.f / C_);
    float var = s2 * (1.f / C_) - mu * mu;
    float inv = rsqrtf(var + 1e-5f);
    bf16* dst = out + (size_t)r * C_;
#pragma unroll
    for (int i = 0; i < 12; ++i) {
        int c = lane + i * 32;
        dst[c] = (bf16)((vals[i] - mu) * inv * w[c] + b[c]);
    }
}

// ---------------------------------------------------------------------------
// Tiled WMMA GEMM:  D[M,N] = A[M,K] (bf16) @ W[N,K]^T (bf16) + bias.
// Block = 8 waves; block tile 256x64. W is staged per 64-deep k-chunk into LDS
// (8 KB) with async global->LDS copies (ASYNCcnt) when the toolchain exposes
// them, so the 8 waves share one copy of the B-tile. Each wave computes a
// 32x64 tile (2 A-frags x 4 B-frags = 8 v_wmma per 32-deep k-step).
// Epilogues: MODE 0 bf16 (QKV) | MODE 1 window-reverse scatter + residual f32
// | MODE 2 exact GELU bf16 (fc1) | MODE 3 + residual(aux) f32 (fc2).
template<int MODE>
__global__ __launch_bounds__(256)
void gemm_wmma(const bf16* __restrict__ A, const bf16* __restrict__ W,
               const float* __restrict__ bias, int M, int N, int K,
               void* __restrict__ outp, const float* __restrict__ aux) {
    __shared__ __align__(32) bf16 Wl[64 * 64];   // staged W chunk [col][k]

    int wave = threadIdx.x >> 5;
    int lane = threadIdx.x & 31;
    int hlf  = lane >> 4;
    int l16  = lane & 15;
    int mBase = blockIdx.y * 256 + wave * 32;
    int nBase = blockIdx.x * 64;

    const bf16* arow0 = A + (size_t)(mBase + l16) * K;
    const bf16* arow1 = A + (size_t)(mBase + 16 + l16) * K;

    // staging assignment: each thread owns (1 col, 16 contiguous k) = 2x16B
    int scol = threadIdx.x >> 2;
    int skl  = (threadIdx.x & 3) * 16;
    const bf16* sgl = W + (size_t)(nBase + scol) * K + skl;
    bf16*       sld = &Wl[scol * 64 + skl];

    int cols[4];
    v8f acc[2][4];
#pragma unroll
    for (int t = 0; t < 4; ++t) {
        cols[t] = nBase + t * 16 + l16;
        float bv = bias[cols[t]];
#pragma unroll
        for (int e = 0; e < 8; ++e) { acc[0][t][e] = bv; acc[1][t][e] = bv; }
    }

    for (int kc = 0; kc < K; kc += 64) {
#if HAVE_ASYNC_LDS
        __builtin_amdgcn_global_load_async_to_lds_b128(
            (as1_v4i*)(sgl + kc),     (as3_v4i*)sld,       0, 0);
        __builtin_amdgcn_global_load_async_to_lds_b128(
            (as1_v4i*)(sgl + kc + 8), (as3_v4i*)(sld + 8), 0, 0);
        __builtin_amdgcn_s_wait_asynccnt(0);
#else
        *(v8bf*)sld       = *(const v8bf*)(sgl + kc);
        *(v8bf*)(sld + 8) = *(const v8bf*)(sgl + kc + 8);
#endif
        __syncthreads();

        if (kc + 64 < K) {   // prefetch next A chunk (global_prefetch_b8)
            __builtin_prefetch(arow0 + kc + 64 + hlf * 8, 0, 1);
            __builtin_prefetch(arow1 + kc + 64 + hlf * 8, 0, 1);
        }
#pragma unroll
        for (int kk = 0; kk < 64; kk += 32) {
            int c0 = kc + kk + hlf * 16;
            v16bf a0 = load_a_frag(arow0, c0);
            v16bf a1 = load_a_frag(arow1, c0);
#pragma unroll
            for (int t = 0; t < 4; ++t) {
                v16bf bfr = *(const v16bf*)(&Wl[(t * 16 + l16) * 64 + kk + hlf * 16]);
                acc[0][t] = __builtin_amdgcn_wmma_f32_16x16x32_bf16(
                    false, a0, false, bfr, (short)0, acc[0][t], false, false);
                acc[1][t] = __builtin_amdgcn_wmma_f32_16x16x32_bf16(
                    false, a1, false, bfr, (short)0, acc[1][t], false, false);
            }
        }
        __syncthreads();
    }

    if constexpr (MODE == 0 || MODE == 2) {
        bf16* out = (bf16*)outp;
#pragma unroll
        for (int u = 0; u < 2; ++u)
#pragma unroll
            for (int t = 0; t < 4; ++t)
#pragma unroll
                for (int e = 0; e < 8; ++e) {
                    int m = mBase + u * 16 + hlf * 8 + e;
                    float v = acc[u][t][e];
                    if constexpr (MODE == 2)
                        v = 0.5f * v * (1.0f + erff(v * 0.70710678118654752f));
                    out[(size_t)m * N + cols[t]] = (bf16)v;
                }
    } else if constexpr (MODE == 1) {
        float* X1 = (float*)outp;
#pragma unroll
        for (int u = 0; u < 2; ++u)
#pragma unroll
            for (int e = 0; e < 8; ++e) {
                int m = mBase + u * 16 + hlf * 8 + e;
                size_t rowo = (size_t)win_to_img_row(m) * N;
#pragma unroll
                for (int t = 0; t < 4; ++t) {
                    size_t o = rowo + cols[t];
                    X1[o] = aux[o] + acc[u][t][e];
                }
            }
    } else { // MODE 3
        float* out = (float*)outp;
#pragma unroll
        for (int u = 0; u < 2; ++u)
#pragma unroll
            for (int e = 0; e < 8; ++e) {
                size_t rowo = (size_t)(mBase + u * 16 + hlf * 8 + e) * N;
#pragma unroll
                for (int t = 0; t < 4; ++t) {
                    size_t o = rowo + cols[t];
                    out[o] = aux[o] + acc[u][t][e];
                }
            }
    }
}

// ---------------------------------------------------------------------------
// Windowed attention, one block per (window, head). 4 waves cover 49 query
// rows (padded to 64). S = q k^T (WMMA) -> *scale + table[bias+mask] ->
// softmax -> P (bf16 via LDS transpose) -> O = P V (WMMA, K=64 in 2 steps).
__global__ __launch_bounds__(128)
void attn_kernel(const bf16* __restrict__ QKV, const float* __restrict__ T,
                 bf16* __restrict__ O) {
    __shared__ __align__(32) bf16 Vt[32 * 64];       // V transposed [d][n], padded cols zeroed
    __shared__ __align__(32) bf16 P[4][16 * 64];     // per-wave probability tiles

    int w = blockIdx.x / HEADS_;
    int h = blockIdx.x - w * HEADS_;
    int tid  = threadIdx.x;
    int wave = tid >> 5, lane = tid & 31, hlf = lane >> 4, l16 = lane & 15;

    size_t wbase = (size_t)w * NTOK * QKV_N;
    int widx = w & 63;
    int cls  = ((widx >> 3) == 7 ? 2 : 0) + ((widx & 7) == 7 ? 1 : 0);
    const float* Tt = T + (size_t)(h * 4 + cls) * 4096;

    // stage V transposed in LDS (zero the 49..63 padding)
    for (int i = tid; i < 32 * 64; i += 128) {
        int n = i & 63, d = i >> 6;
        bf16 v = (bf16)0.0f;
        if (n < NTOK) v = QKV[wbase + (size_t)n * QKV_N + h * HD_ + d];
        Vt[d * 64 + n] = v;
    }
    __syncthreads();

    int mBase = wave * 16;

    // ---- S = q k^T ----
    int mA = mBase + l16; if (mA > 48) mA = 48;      // avoid OOB reads only
    const bf16* qrow = QKV + wbase + (size_t)mA * QKV_N + C_ + h * HD_;
    v16bf aq = load_a_frag(qrow, hlf * 16);

    v8f acc[4];
#pragma unroll
    for (int t = 0; t < 4; ++t) {
        int n = t * 16 + l16; if (n > 48) n = 48;
        v16bf bk = *(const v16bf*)(QKV + wbase + (size_t)n * QKV_N + 2 * C_ + h * HD_ + hlf * 16);
        v8f z = {0.f, 0.f, 0.f, 0.f, 0.f, 0.f, 0.f, 0.f};
        acc[t] = __builtin_amdgcn_wmma_f32_16x16x32_bf16(
            false, aq, false, bk, (short)0, z, false, false);
    }

    // ---- scale + precomputed bias+mask table ----
    const float SCALE = 0.17677669529663687f;   // 32^-0.5
#pragma unroll
    for (int t = 0; t < 4; ++t) {
        int n = t * 16 + l16;
#pragma unroll
        for (int e = 0; e < 8; ++e) {
            int m = mBase + hlf * 8 + e;
            acc[t][e] = acc[t][e] * SCALE + Tt[m * 64 + n];
        }
    }

    // ---- softmax along the row (16 lanes of this half x 4 tiles) ----
#pragma unroll
    for (int e = 0; e < 8; ++e) {
        float mx = fmaxf(fmaxf(acc[0][e], acc[1][e]), fmaxf(acc[2][e], acc[3][e]));
#pragma unroll
        for (int msk = 1; msk < 16; msk <<= 1) mx = fmaxf(mx, __shfl_xor(mx, msk, 32));
        float sum = 0.f;
#pragma unroll
        for (int t = 0; t < 4; ++t) { float p = __expf(acc[t][e] - mx); acc[t][e] = p; sum += p; }
#pragma unroll
        for (int msk = 1; msk < 16; msk <<= 1) sum += __shfl_xor(sum, msk, 32);
        float inv = 1.f / sum;
        int mloc = hlf * 8 + e;
#pragma unroll
        for (int t = 0; t < 4; ++t)
            P[wave][mloc * 64 + t * 16 + l16] = (bf16)(acc[t][e] * inv);
    }
    __syncthreads();

    // ---- O = P V ----
    v8f acc2[2];
#pragma unroll
    for (int t = 0; t < 2; ++t) {
        v8f z = {0.f, 0.f, 0.f, 0.f, 0.f, 0.f, 0.f, 0.f};
        acc2[t] = z;
    }
    const bf16* prow = &P[wave][l16 * 64];
#pragma unroll
    for (int ks = 0; ks < 2; ++ks) {
        v16bf ap = load_a_frag(prow, ks * 32 + hlf * 16);
#pragma unroll
        for (int t = 0; t < 2; ++t) {
            int d = t * 16 + l16;
            v16bf bv = *(const v16bf*)(&Vt[d * 64 + ks * 32 + hlf * 16]);
            acc2[t] = __builtin_amdgcn_wmma_f32_16x16x32_bf16(
                false, ap, false, bv, (short)0, acc2[t], false, false);
        }
    }
#pragma unroll
    for (int t = 0; t < 2; ++t)
#pragma unroll
        for (int e = 0; e < 8; ++e) {
            int m = mBase + hlf * 8 + e;
            if (m < NTOK)
                O[((size_t)w * NTOK + m) * C_ + h * HD_ + t * 16 + l16] = (bf16)acc2[t][e];
        }
}

// ---------------------------------------------------------------------------
extern "C" void kernel_launch(void* const* d_in, const int* in_sizes, int n_in,
                              void* d_out, int out_size, void* d_ws, size_t ws_size,
                              hipStream_t stream) {
    (void)in_sizes; (void)n_in; (void)out_size; (void)ws_size;

    const float* x       = (const float*)d_in[0];
    const float* norm1_w = (const float*)d_in[1];
    const float* norm1_b = (const float*)d_in[2];
    const float* qkv_w   = (const float*)d_in[3];
    const float* qkv_b   = (const float*)d_in[4];
    const float* rb      = (const float*)d_in[5];
    const float* proj_w  = (const float*)d_in[6];
    const float* proj_b  = (const float*)d_in[7];
    const float* norm2_w = (const float*)d_in[8];
    const float* norm2_b = (const float*)d_in[9];
    const float* fc1_w   = (const float*)d_in[10];
    const float* fc1_b   = (const float*)d_in[11];
    const float* fc2_w   = (const float*)d_in[12];
    const float* fc2_b   = (const float*)d_in[13];

    char* ws = (char*)d_ws;
    // region 0: bf16 [NROWS x 384] — T (ln1 out) -> O (attn out) -> H (ln2 out)
    bf16* R0 = (bf16*)ws;
    size_t off1 = (size_t)NROWS * C_ * 2;                 // 77,070,336
    // region 1: bf16 QKV [NROWS x 1152], later reused as f32 X1 [NROWS x 384]
    char* R1 = ws + off1;
    bf16*  QKV = (bf16*)R1;
    float* X1  = (float*)R1;
    size_t off2 = off1 + (size_t)NROWS * QKV_N * 2;       // +231,211,008
    // region 2: bf16 GELU activations [NROWS x 1536]
    bf16* G = (bf16*)(ws + off2);
    size_t off3 = off2 + (size_t)NROWS * MLPH * 2;        // +308,281,344
    // region 3: packed bf16 weights + attention tables
    bf16* qkvw = (bf16*)(ws + off3);
    bf16* prjw = qkvw + QKV_N * C_;
    bf16* f1w  = prjw + C_ * C_;
    bf16* f2w  = f1w  + MLPH * C_;
    float* Tab = (float*)(f2w + C_ * MLPH);               // 12*4*64*64 f32

    // 1) pack weights + build bias/mask tables
    pack_bf16<<<(QKV_N * C_ + 255) / 256, 256, 0, stream>>>(qkv_w,  qkvw, QKV_N * C_);
    pack_bf16<<<(C_ * C_    + 255) / 256, 256, 0, stream>>>(proj_w, prjw, C_ * C_);
    pack_bf16<<<(MLPH * C_  + 255) / 256, 256, 0, stream>>>(fc1_w,  f1w,  MLPH * C_);
    pack_bf16<<<(C_ * MLPH  + 255) / 256, 256, 0, stream>>>(fc2_w,  f2w,  C_ * MLPH);
    build_attn_tables<<<HEADS_ * 4, 256, 0, stream>>>(rb, Tab);

    // 2) LN1 + shift + window partition -> R0 (bf16, window-row order)
    ln_kernel<true><<<NROWS / 8, 256, 0, stream>>>(x, norm1_w, norm1_b, R0);

    // 3) QKV GEMM: [100352,384] @ [384,1152]
    gemm_wmma<0><<<dim3(QKV_N / 64, NROWS / 256), 256, 0, stream>>>(
        R0, qkvw, qkv_b, NROWS, QKV_N, C_, (void*)QKV, nullptr);

    // 4) windowed attention -> R0 (reused, attn output bf16)
    attn_kernel<<<(NROWS / NTOK) * HEADS_, 128, 0, stream>>>(QKV, Tab, R0);

    // 5) proj GEMM + window-reverse/roll scatter + residual -> X1 (f32)
    gemm_wmma<1><<<dim3(C_ / 64, NROWS / 256), 256, 0, stream>>>(
        R0, prjw, proj_b, NROWS, C_, C_, (void*)X1, x);

    // 6) LN2 -> R0 (bf16)
    ln_kernel<false><<<NROWS / 8, 256, 0, stream>>>(X1, norm2_w, norm2_b, R0);

    // 7) fc1 GEMM + GELU -> G (bf16)
    gemm_wmma<2><<<dim3(MLPH / 64, NROWS / 256), 256, 0, stream>>>(
        R0, f1w, fc1_b, NROWS, MLPH, C_, (void*)G, nullptr);

    // 8) fc2 GEMM + residual(X1) -> d_out (f32)
    gemm_wmma<3><<<dim3(C_ / 64, NROWS / 256), 256, 0, stream>>>(
        G, f2w, fc2_b, NROWS, C_, MLPH, d_out, X1);
}
